// Attention_28252294873693
// MI455X (gfx1250) — matrix operand order repaired and checked by reading, not compiled, not run on hardware
//
#include <hip/hip_runtime.h>

#define BN   4
#define SEQ  2048
#define CH   1024
#define HD   64
#define NH   16
#define QKV_ELEMS (BN * NH * SEQ * HD)   // 8388608 elements per tensor

#ifndef __has_builtin
#define __has_builtin(x) 0
#endif

#if __has_builtin(__builtin_amdgcn_cvt_pk_bf16_f32)
#define HAVE_PK_BF16 1
#else
#define HAVE_PK_BF16 0
#endif

#if __has_builtin(__builtin_amdgcn_global_load_async_to_lds_b128)
#define HAVE_ASYNC_LDS 1
#else
#define HAVE_ASYNC_LDS 0
#endif

#define GLOBAL_AS __attribute__((address_space(1)))
#define LDS_AS    __attribute__((address_space(3)))

typedef __attribute__((ext_vector_type(16))) __bf16 v16bf;
typedef __attribute__((ext_vector_type(8)))  float  v8f;
typedef __attribute__((ext_vector_type(4)))  int    v4i;

union Frag16 {
  v16bf          v;
  unsigned short u16[16];
  unsigned int   u32[8];
  uint4          u128[2];
};

union FragAcc {
  v8f   v;
  float f[8];
};

__device__ __forceinline__ unsigned short f2bf_manual(float x) {
  unsigned int u = __float_as_uint(x);
  u += 0x7FFFu + ((u >> 16) & 1u);
  return (unsigned short)(u >> 16);
}

__device__ __forceinline__ unsigned int bfpack2(float lo, float hi) {
#if HAVE_PK_BF16
  auto r = __builtin_amdgcn_cvt_pk_bf16_f32(lo, hi);
  unsigned int u;
  __builtin_memcpy(&u, &r, 4);
  return u;
#else
  return (unsigned)f2bf_manual(lo) | ((unsigned)f2bf_manual(hi) << 16);
#endif
}

__device__ __forceinline__ unsigned short bf16s(float x) {
  return (unsigned short)(bfpack2(x, x) & 0xFFFFu);
}

#if HAVE_ASYNC_LDS
typedef GLOBAL_AS v4i* gv4i_p;
typedef LDS_AS    v4i* lv4i_p;
__device__ __forceinline__ void async_b128(const void* g, void* l) {
  __builtin_amdgcn_global_load_async_to_lds_b128(
      (gv4i_p)(unsigned long long)g, (lv4i_p)l, 0, 0);
}
#if __has_builtin(__builtin_amdgcn_s_wait_asynccnt)
#define WAIT_ASYNC(n) __builtin_amdgcn_s_wait_asynccnt(n)
#else
#define WAIT_ASYNC(n) asm volatile("s_wait_asynccnt %0" ::"i"(n) : "memory")
#endif
#endif

__device__ __forceinline__ v8f wmma_bf16(const Frag16& a, const Frag16& b, v8f c) {
  return __builtin_amdgcn_wmma_f32_16x16x32_bf16(false, a.v, false, b.v,
                                                 (short)0, c, false, false);
}

// ---------------------------------------------------------------------------
// Kernel 0: one-pass fp32 -> bf16 conversion (8 elements / thread)
// ---------------------------------------------------------------------------
__global__ __launch_bounds__(256) void f32_to_bf16_kernel(
    const float* __restrict__ src, unsigned short* __restrict__ dst) {
  size_t i = ((size_t)blockIdx.x * 256 + threadIdx.x) * 8;
  float4 a0 = *(const float4*)(src + i);
  float4 a1 = *(const float4*)(src + i + 4);
  uint4 p;
  p.x = bfpack2(a0.x, a0.y);
  p.y = bfpack2(a0.z, a0.w);
  p.z = bfpack2(a1.x, a1.y);
  p.w = bfpack2(a1.z, a1.w);
  *(uint4*)(dst + i) = p;
}

// ---------------------------------------------------------------------------
// Kernel 1: qkv = xb @ Wb^T (both bf16) -> q,k,v in [B,H,N,hd] bf16.
// 128(M) x 64(F) block tile, 8 waves each 32x32 (4 wmma / k-step).
// Double-buffered LDS, async-to-LDS pipelined with s_wait_asynccnt.
// ---------------------------------------------------------------------------
__global__ __launch_bounds__(256) void qkv_gemm_kernel(
    const unsigned short* __restrict__ Xb, const unsigned short* __restrict__ Wb,
    unsigned short* __restrict__ qkv) {
  __shared__ unsigned short sA[2][128 * 40];
  __shared__ unsigned short sB[2][64 * 40];

  const int tid  = threadIdx.x;
  const int lane = tid & 31;
  const int wv   = tid >> 5;
  const int m0   = blockIdx.x * 128;
  const int f0   = blockIdx.y * 64;
  const int mrow = (wv & 3) * 32;
  const int fcol = (wv >> 2) * 32;

  FragAcc acc[4];
#pragma unroll
  for (int t = 0; t < 4; ++t)
#pragma unroll
    for (int j = 0; j < 8; ++j) acc[t].f[j] = 0.f;

  const int arow = tid >> 1;        // 0..127
  const int acg  = (tid & 1) * 16;  // 0 or 16
  const int brow = tid >> 2;        // 0..63
  const int bcg  = (tid & 3) * 8;   // 0,8,16,24

  // stage tile at k0 into LDS buffer `buf` (3 async b128 per thread)
  auto issue = [&](int k0, int buf) {
    const unsigned short* srcA = Xb + (size_t)(m0 + arow) * CH + k0 + acg;
    const unsigned short* srcB = Wb + (size_t)(f0 + brow) * CH + k0 + bcg;
#if HAVE_ASYNC_LDS
    async_b128(srcA,     &sA[buf][arow * 40 + acg]);
    async_b128(srcA + 8, &sA[buf][arow * 40 + acg + 8]);
    async_b128(srcB,     &sB[buf][brow * 40 + bcg]);
#else
    *(uint4*)(&sA[buf][arow * 40 + acg])     = *(const uint4*)(srcA);
    *(uint4*)(&sA[buf][arow * 40 + acg + 8]) = *(const uint4*)(srcA + 8);
    *(uint4*)(&sB[buf][brow * 40 + bcg])     = *(const uint4*)(srcB);
#endif
  };

  issue(0, 0);
  for (int k0 = 0, t = 0; k0 < CH; k0 += 32, ++t) {
    const int cur  = t & 1;
    const bool more = (k0 + 32 < CH);
    if (more) issue(k0 + 32, cur ^ 1);   // prefetch next tile into other buffer
#if HAVE_ASYNC_LDS
    if (more) WAIT_ASYNC(3);             // retire current tile's 3 loads
    else      WAIT_ASYNC(0);
#endif
    __syncthreads();

    Frag16 af0, af1;
    {
      int r  = mrow + (lane & 15);
      int kb = (lane >> 4) * 16;
      af0.u128[0] = *(const uint4*)(&sA[cur][r * 40 + kb]);
      af0.u128[1] = *(const uint4*)(&sA[cur][r * 40 + 8 + kb]);
      af1.u128[0] = *(const uint4*)(&sA[cur][(r + 16) * 40 + kb]);
      af1.u128[1] = *(const uint4*)(&sA[cur][(r + 16) * 40 + 8 + kb]);
    }
    Frag16 bf0, bf1;
    {
      int kb  = (lane >> 4) * 16;
      int fl0 = fcol + (lane & 15);
      int fl1 = fl0 + 16;
#pragma unroll
      for (int j = 0; j < 8; ++j) {
        bf0.u32[j] = *(const unsigned int*)(&sB[cur][fl0 * 40 + kb + 2 * j]);
        bf1.u32[j] = *(const unsigned int*)(&sB[cur][fl1 * 40 + kb + 2 * j]);
      }
    }
    acc[0].v = wmma_bf16(af0, bf0, acc[0].v);
    acc[1].v = wmma_bf16(af0, bf1, acc[1].v);
    acc[2].v = wmma_bf16(af1, bf0, acc[2].v);
    acc[3].v = wmma_bf16(af1, bf1, acc[3].v);
    __syncthreads();   // all waves done with `cur` before it is re-filled
  }

  // epilogue: scatter into q/k/v [B,H,N,hd] bf16
#pragma unroll
  for (int ar = 0; ar < 2; ++ar) {
#pragma unroll
    for (int c = 0; c < 2; ++c) {
      const FragAcc& a = acc[ar * 2 + c];
      int fg = f0 + fcol + c * 16 + (lane & 15);
      int t  = fg >> 10;            // 0=q 1=k 2=v
      int h  = (fg >> 6) & (NH - 1);
      int d  = fg & (HD - 1);
      int mg0 = m0 + mrow + ar * 16 + (lane >> 4) * 8;
      int b   = mg0 >> 11;
      int n0r = mg0 & (SEQ - 1);
      unsigned short* dst = qkv + (size_t)t * QKV_ELEMS +
          (((size_t)(b * NH + h) * SEQ + n0r) * HD + d);
#pragma unroll
      for (int j = 0; j < 8; ++j) dst[j * HD] = bf16s(a.f[j]);
    }
  }
}

// ---------------------------------------------------------------------------
// Kernel 2: flash attention per (b,h). Block: 8 waves x 16 query rows = 128.
// 64-key tiles of K/V via LDS (K async); online softmax; 16 wmma / key tile.
// ---------------------------------------------------------------------------
__global__ __launch_bounds__(256) void attn_kernel(
    const unsigned short* __restrict__ Q,
    const unsigned short* __restrict__ K,
    const unsigned short* __restrict__ V,
    const float* __restrict__ bias,
    unsigned short* __restrict__ Out) {
  __shared__ unsigned short kbuf[64 * 72];        // [m][d], padded stride
  __shared__ unsigned short vt[64 * 68];          // [d][m] transposed
  __shared__ unsigned short pbuf[8 * 16 * 64];    // per-wave P scratch

  const int tid  = threadIdx.x;
  const int lane = tid & 31;
  const int wv   = tid >> 5;
  const int bh   = blockIdx.x;                    // b*16 + h
  const int n0   = blockIdx.y * 128 + wv * 16;
  const float scale = 0.125f;                     // hd^-0.5

  const unsigned short* qbase = Q + (size_t)bh * SEQ * HD;
  const unsigned short* kbase = K + (size_t)bh * SEQ * HD;
  const unsigned short* vbase = V + (size_t)bh * SEQ * HD;

  Frag16 qf0, qf1;
  {
    int r  = n0 + (lane & 15);
    int db = (lane >> 4) * 16;
    const unsigned short* qp = qbase + (size_t)r * HD;
    qf0.u128[0] = *(const uint4*)(qp + db);
    qf0.u128[1] = *(const uint4*)(qp + 8 + db);
    qf1.u128[0] = *(const uint4*)(qp + 32 + db);
    qf1.u128[1] = *(const uint4*)(qp + 40 + db);
  }

  FragAcc oacc[4];
#pragma unroll
  for (int c = 0; c < 4; ++c)
#pragma unroll
    for (int j = 0; j < 8; ++j) oacc[c].f[j] = 0.f;
  float mrowv[8], lrowv[8];
#pragma unroll
  for (int j = 0; j < 8; ++j) { mrowv[j] = -3.0e38f; lrowv[j] = 0.f; }

  const int kb = (lane >> 4) * 16;

  for (int m0 = 0; m0 < SEQ; m0 += 64) {
#pragma unroll
    for (int s = 0; s < 2; ++s) {
      int c   = tid + s * 256;
      int row = c >> 3;             // 0..63
      int dg  = (c & 7) * 8;        // 0..56
      const unsigned short* kg = kbase + (size_t)(m0 + row) * HD + dg;
#if HAVE_ASYNC_LDS
      async_b128(kg, &kbuf[row * 72 + dg]);
#else
      *(uint4*)(&kbuf[row * 72 + dg]) = *(const uint4*)kg;
#endif
      uint4 vv = *(const uint4*)(vbase + (size_t)(m0 + row) * HD + dg);
      const unsigned short* pv = (const unsigned short*)&vv;
#pragma unroll
      for (int ii = 0; ii < 8; ++ii) vt[(dg + ii) * 68 + row] = pv[ii];
    }
    if (m0 + 64 < SEQ) {
      const size_t nx = (size_t)(m0 + 64 + (tid & 63)) * HD;
      __builtin_prefetch(kbase + nx, 0, 1);
      __builtin_prefetch(vbase + nx, 0, 1);
    }
#if HAVE_ASYNC_LDS
    WAIT_ASYNC(0);
#endif
    __syncthreads();

    // S = Q K^T (16 queries x 64 keys): 8 wmma
    FragAcc sfr[4];
#pragma unroll
    for (int c = 0; c < 4; ++c) {
#pragma unroll
      for (int j = 0; j < 8; ++j) sfr[c].f[j] = 0.f;
      Frag16 kf0, kf1;
      int m_l = c * 16 + (lane & 15);
#pragma unroll
      for (int j = 0; j < 8; ++j) {
        kf0.u32[j] = *(const unsigned int*)(&kbuf[m_l * 72 + kb + 2 * j]);
        kf1.u32[j] = *(const unsigned int*)(&kbuf[m_l * 72 + 32 + kb + 2 * j]);
      }
      sfr[c].v = wmma_bf16(qf0, kf0, sfr[c].v);
      sfr[c].v = wmma_bf16(qf1, kf1, sfr[c].v);
    }

#pragma unroll
    for (int j = 0; j < 8; ++j) {
      int n_g = n0 + (lane >> 4) * 8 + j;
      const float* bp = bias + (size_t)n_g * SEQ + m0 + (lane & 15);
#pragma unroll
      for (int c = 0; c < 4; ++c)
        sfr[c].f[j] = sfr[c].f[j] * scale + bp[c * 16];
    }

    unsigned short* pw = &pbuf[wv * 16 * 64];
#pragma unroll
    for (int j = 0; j < 8; ++j) {
      float mx = fmaxf(fmaxf(sfr[0].f[j], sfr[1].f[j]),
                       fmaxf(sfr[2].f[j], sfr[3].f[j]));
#pragma unroll
      for (int off = 1; off < 16; off <<= 1)
        mx = fmaxf(mx, __shfl_xor(mx, off, 32));
      float mnew  = fmaxf(mrowv[j], mx);
      float alpha = __expf(mrowv[j] - mnew);
      float p[4];
      float ls = 0.f;
#pragma unroll
      for (int c = 0; c < 4; ++c) { p[c] = __expf(sfr[c].f[j] - mnew); ls += p[c]; }
#pragma unroll
      for (int off = 1; off < 16; off <<= 1)
        ls += __shfl_xor(ls, off, 32);
      lrowv[j] = lrowv[j] * alpha + ls;
      mrowv[j] = mnew;
      oacc[0].f[j] *= alpha; oacc[1].f[j] *= alpha;
      oacc[2].f[j] *= alpha; oacc[3].f[j] *= alpha;
      int row = (lane >> 4) * 8 + j;
#pragma unroll
      for (int c = 0; c < 4; ++c)
        pw[row * 64 + c * 16 + (lane & 15)] = bf16s(p[c]);
    }
    __syncthreads();

    Frag16 pf0, pf1;
    {
      int r   = lane & 15;
      int kb2 = (lane >> 4) * 16;
      pf0.u128[0] = *(const uint4*)(&pw[r * 64 + kb2]);
      pf0.u128[1] = *(const uint4*)(&pw[r * 64 + 8 + kb2]);
      pf1.u128[0] = *(const uint4*)(&pw[r * 64 + 32 + kb2]);
      pf1.u128[1] = *(const uint4*)(&pw[r * 64 + 40 + kb2]);
    }
#pragma unroll
    for (int c2 = 0; c2 < 4; ++c2) {
      Frag16 vf0, vf1;
      int d_l = c2 * 16 + (lane & 15);
#pragma unroll
      for (int j = 0; j < 8; ++j) {
        vf0.u32[j] = *(const unsigned int*)(&vt[d_l * 68 + kb + 2 * j]);
        vf1.u32[j] = *(const unsigned int*)(&vt[d_l * 68 + 32 + kb + 2 * j]);
      }
      oacc[c2].v = wmma_bf16(pf0, vf0, oacc[c2].v);
      oacc[c2].v = wmma_bf16(pf1, vf1, oacc[c2].v);
    }
    __syncthreads();
  }

  const int b = bh >> 4, h = bh & (NH - 1);
#pragma unroll
  for (int j = 0; j < 8; ++j) {
    int n_g     = n0 + (lane >> 4) * 8 + j;
    float inv   = 1.0f / lrowv[j];
    size_t base = ((size_t)b * SEQ + n_g) * CH + (size_t)h * HD;
#pragma unroll
    for (int c2 = 0; c2 < 4; ++c2)
      Out[base + c2 * 16 + (lane & 15)] = bf16s(oacc[c2].f[j] * inv);
  }
}

// ---------------------------------------------------------------------------
// Kernel 3: out = ao @ Wproj^T + b_proj (bf16 x bf16 -> fp32 out)
// Same pipelined double-buffered structure as kernel 1.
// ---------------------------------------------------------------------------
__global__ __launch_bounds__(256) void proj_gemm_kernel(
    const unsigned short* __restrict__ A, const unsigned short* __restrict__ Wb,
    const float* __restrict__ bvec, float* __restrict__ out) {
  __shared__ unsigned short sA[2][128 * 40];
  __shared__ unsigned short sB[2][64 * 40];

  const int tid  = threadIdx.x;
  const int lane = tid & 31;
  const int wv   = tid >> 5;
  const int m0   = blockIdx.x * 128;
  const int f0   = blockIdx.y * 64;
  const int mrow = (wv & 3) * 32;
  const int fcol = (wv >> 2) * 32;

  FragAcc acc[4];
#pragma unroll
  for (int t = 0; t < 4; ++t)
#pragma unroll
    for (int j = 0; j < 8; ++j) acc[t].f[j] = 0.f;

  const int arow = tid >> 1;
  const int acg  = (tid & 1) * 16;
  const int brow = tid >> 2;
  const int bcg  = (tid & 3) * 8;

  auto issue = [&](int k0, int buf) {
    const unsigned short* srcA = A  + (size_t)(m0 + arow) * CH + k0 + acg;
    const unsigned short* srcB = Wb + (size_t)(f0 + brow) * CH + k0 + bcg;
#if HAVE_ASYNC_LDS
    async_b128(srcA,     &sA[buf][arow * 40 + acg]);
    async_b128(srcA + 8, &sA[buf][arow * 40 + acg + 8]);
    async_b128(srcB,     &sB[buf][brow * 40 + bcg]);
#else
    *(uint4*)(&sA[buf][arow * 40 + acg])     = *(const uint4*)(srcA);
    *(uint4*)(&sA[buf][arow * 40 + acg + 8]) = *(const uint4*)(srcA + 8);
    *(uint4*)(&sB[buf][brow * 40 + bcg])     = *(const uint4*)(srcB);
#endif
  };

  issue(0, 0);
  for (int k0 = 0, t = 0; k0 < CH; k0 += 32, ++t) {
    const int cur   = t & 1;
    const bool more = (k0 + 32 < CH);
    if (more) issue(k0 + 32, cur ^ 1);
#if HAVE_ASYNC_LDS
    if (more) WAIT_ASYNC(3);
    else      WAIT_ASYNC(0);
#endif
    __syncthreads();

    Frag16 af0, af1;
    {
      int r  = mrow + (lane & 15);
      int kb = (lane >> 4) * 16;
      af0.u128[0] = *(const uint4*)(&sA[cur][r * 40 + kb]);
      af0.u128[1] = *(const uint4*)(&sA[cur][r * 40 + 8 + kb]);
      af1.u128[0] = *(const uint4*)(&sA[cur][(r + 16) * 40 + kb]);
      af1.u128[1] = *(const uint4*)(&sA[cur][(r + 16) * 40 + 8 + kb]);
    }
    Frag16 bf0, bf1;
    {
      int kb  = (lane >> 4) * 16;
      int fl0 = fcol + (lane & 15);
      int fl1 = fl0 + 16;
#pragma unroll
      for (int j = 0; j < 8; ++j) {
        bf0.u32[j] = *(const unsigned int*)(&sB[cur][fl0 * 40 + kb + 2 * j]);
        bf1.u32[j] = *(const unsigned int*)(&sB[cur][fl1 * 40 + kb + 2 * j]);
      }
    }
    acc[0].v = wmma_bf16(af0, bf0, acc[0].v);
    acc[1].v = wmma_bf16(af0, bf1, acc[1].v);
    acc[2].v = wmma_bf16(af1, bf0, acc[2].v);
    acc[3].v = wmma_bf16(af1, bf1, acc[3].v);
    __syncthreads();
  }

#pragma unroll
  for (int ar = 0; ar < 2; ++ar) {
#pragma unroll
    for (int c = 0; c < 2; ++c) {
      const FragAcc& a = acc[ar * 2 + c];
      int fg   = f0 + fcol + c * 16 + (lane & 15);
      float bb = bvec[fg];
      int mg0  = m0 + mrow + ar * 16 + (lane >> 4) * 8;
      float* dst = out + (size_t)mg0 * CH + fg;
#pragma unroll
      for (int j = 0; j < 8; ++j) dst[(size_t)j * CH] = a.f[j] + bb;
    }
  }
}

// ---------------------------------------------------------------------------
extern "C" void kernel_launch(void* const* d_in, const int* in_sizes, int n_in,
                              void* d_out, int out_size, void* d_ws, size_t ws_size,
                              hipStream_t stream) {
  (void)in_sizes; (void)n_in; (void)out_size; (void)ws_size;
  const float* x     = (const float*)d_in[0];
  const float* Wqkv  = (const float*)d_in[1];
  const float* Wproj = (const float*)d_in[2];
  const float* bproj = (const float*)d_in[3];
  const float* gbias = (const float*)d_in[4];
  float* out = (float*)d_out;

  unsigned short* ws = (unsigned short*)d_ws;
  unsigned short* q   = ws;                          // [B,H,N,hd] bf16
  unsigned short* k   = ws + (size_t)QKV_ELEMS;
  unsigned short* v   = ws + (size_t)2 * QKV_ELEMS;
  unsigned short* ao  = ws + (size_t)3 * QKV_ELEMS;  // [B,N,C] bf16
  unsigned short* xb  = ws + (size_t)4 * QKV_ELEMS;  // x in bf16
  unsigned short* wqb = xb + (size_t)BN * SEQ * CH;  // W_qkv in bf16
  unsigned short* wpb = wqb + (size_t)3 * CH * CH;   // W_proj in bf16

  // 0) one-pass fp32 -> bf16 conversions (removes all conversion VALU from GEMMs)
  f32_to_bf16_kernel<<<(BN * SEQ * CH) / 2048, 256, 0, stream>>>(x, xb);
  f32_to_bf16_kernel<<<(3 * CH * CH) / 2048, 256, 0, stream>>>(Wqkv, wqb);
  f32_to_bf16_kernel<<<(CH * CH) / 2048, 256, 0, stream>>>(Wproj, wpb);

  // 1) QKV projection: M=B*N=8192 tiles of 128, F=3C=3072 tiles of 64
  qkv_gemm_kernel<<<dim3(64, 48), 256, 0, stream>>>(xb, wqb, q);
  // 2) attention: (b*h) x (query slabs of 128)
  attn_kernel<<<dim3(BN * NH, SEQ / 128), 256, 0, stream>>>(q, k, v, gbias, ao);
  // 3) output projection + bias
  proj_gemm_kernel<<<dim3(64, 16), 256, 0, stream>>>(ao, wpb, bproj, out);
}
